// VectorQuantize_85564338471333
// MI455X (gfx1250) — matrix operand order repaired and checked by reading, not compiled, not run on hardware
//
#include <hip/hip_runtime.h>
#include <hip/hip_bf16.h>

typedef __bf16 bf16_t;
typedef __attribute__((ext_vector_type(16))) __bf16 v16bf;
typedef __attribute__((ext_vector_type(8)))  __bf16 bf16x8;
typedef __attribute__((ext_vector_type(8)))  float   v8f;
typedef __attribute__((ext_vector_type(4)))  int     v4i;

#define B_      8
#define C_IN_   1024
#define T_      4096
#define D_CB_   256
#define K_CB_   8192
#define N_TOK_  (B_ * T_)            /* 32768 */
#define OUT_ELEMS_ ((size_t)B_ * C_IN_ * T_)
#define LDS_PITCH_ 264               /* 256 + 8 bf16 pad: 528B row -> 4-bank skew */

#if __has_builtin(__builtin_amdgcn_global_load_async_to_lds_b128)
#define ASYNC_LDS 1
#endif

/* 16-byte global -> LDS copy; async (ASYNCcnt-tracked) when available */
__device__ __forceinline__ void copy16_g2l(const void* g, void* l) {
#ifdef ASYNC_LDS
  __builtin_amdgcn_global_load_async_to_lds_b128(
      (__attribute__((address_space(1))) v4i*)(const_cast<void*>(g)),
      (__attribute__((address_space(3))) v4i*)l, 0, 0);
#else
  *(uint4*)l = *(const uint4*)g;
#endif
}

__device__ __forceinline__ void wait_async_copies() {
#ifdef ASYNC_LDS
#if __has_builtin(__builtin_amdgcn_s_wait_asynccnt)
  __builtin_amdgcn_s_wait_asynccnt(0);
#else
  asm volatile("s_wait_asynccnt 0x0" ::: "memory");
#endif
#endif
}

/* ---- fragment helpers (wave32 WMMA operands in this code's contiguous k order, not the ISA table order) ----
 * A 16x32 bf16: lane m=l%16; elems[0..15]=K (l<16?0:16)+i  (contiguous)
 * B 32x16 bf16: lane n=l%16; elems[0..15]=K (l<16?0:16)+i  (contiguous)        */
__device__ __forceinline__ v16bf load_fragA(const bf16_t* rowptr, int k0, int lane) {
  const int kb = k0 + ((lane & 16) ? 16 : 0);
  const bf16x8 lo = *(const bf16x8*)(rowptr + kb);
  const bf16x8 hi = *(const bf16x8*)(rowptr + kb + 8);
  v16bf a;
#pragma unroll
  for (int i = 0; i < 8; ++i) { a[i] = lo[i]; a[i + 8] = hi[i]; }
  return a;
}

__device__ __forceinline__ v16bf load16_contig(const bf16_t* p) {
  const bf16x8 lo = *(const bf16x8*)p;
  const bf16x8 hi = *(const bf16x8*)(p + 8);
  v16bf r;
#pragma unroll
  for (int i = 0; i < 8; ++i) { r[i] = lo[i]; r[i + 8] = hi[i]; }
  return r;
}

/* ======================= prep: weight-norm + bf16 convert ================== */
__global__ void __launch_bounds__(256)
prep_kernel(const float* __restrict__ in_v, const float* __restrict__ in_g,
            const float* __restrict__ out_v, const float* __restrict__ out_g,
            const float* __restrict__ codebook,
            bf16_t* __restrict__ w_in_bf, bf16_t* __restrict__ w_out_bf,
            bf16_t* __restrict__ cb_bf,   bf16_t* __restrict__ cn_bf) {
  __shared__ float red[256];
  const int row = blockIdx.x;
  const int tid = threadIdx.x;
  const float* src;
  int len;
  if (row < D_CB_)               { src = in_v + (size_t)row * C_IN_;                 len = C_IN_; }
  else if (row < D_CB_ + C_IN_)  { src = out_v + (size_t)(row - D_CB_) * D_CB_;      len = D_CB_; }
  else                           { src = codebook + (size_t)(row - D_CB_ - C_IN_) * D_CB_; len = D_CB_; }
  float ss = 0.f;
  for (int i = tid; i < len; i += 256) { float v = src[i]; ss += v * v; }
  red[tid] = ss;
  __syncthreads();
  for (int s = 128; s > 0; s >>= 1) { if (tid < s) red[tid] += red[tid + s]; __syncthreads(); }
  const float norm = sqrtf(red[0]);
  if (row < D_CB_) {
    const float sc = in_g[row] / norm;
    for (int i = tid; i < len; i += 256) w_in_bf[(size_t)row * C_IN_ + i] = (bf16_t)(src[i] * sc);
  } else if (row < D_CB_ + C_IN_) {
    const int c = row - D_CB_;
    const float sc = out_g[c] / norm;
    for (int i = tid; i < len; i += 256) w_out_bf[(size_t)c * D_CB_ + i] = (bf16_t)(src[i] * sc);
  } else {
    const int k = row - D_CB_ - C_IN_;
    const float inv = 1.0f / fmaxf(norm, 1e-12f);
    for (int i = tid; i < len; i += 256) {
      const float v = src[i];
      cb_bf[(size_t)k * D_CB_ + i] = (bf16_t)v;
      cn_bf[(size_t)k * D_CB_ + i] = (bf16_t)(v * inv);
    }
  }
}

/* ============ in_proj: enc[n,d] = w_in @ z + b, stored bf16 [N,256] ========= */
__global__ void __launch_bounds__(256)
inproj_kernel(const float* __restrict__ z, const bf16_t* __restrict__ w_in_bf,
              const float* __restrict__ in_b, bf16_t* __restrict__ enc_bf) {
  const int wid = threadIdx.x >> 5, lane = threadIdx.x & 31;
  const int tile  = blockIdx.x * 8 + wid;
  const int dtile = (tile & 15) << 4;      /* 16 D-tiles  */
  const int ntile = (tile >> 4) << 4;      /* 2048 N-tiles */
  const int lm = lane & 15;
  const int n  = ntile + lm;
  const int b  = n >> 12, t = n & (T_ - 1);
  const float*  zcol = z + ((size_t)b * C_IN_) * T_ + t;
  const bf16_t* arow = w_in_bf + (size_t)(dtile + lm) * C_IN_;
  const int koffB = (lane & 16) ? 16 : 0;
  v8f acc = {};
  for (int k0 = 0; k0 < C_IN_; k0 += 32) {
    const v16bf a = load_fragA(arow, k0, lane);
    v16bf bm;
#pragma unroll
    for (int i = 0; i < 16; ++i)
      bm[i] = (bf16_t)zcol[(size_t)(k0 + koffB + i) * T_];
    acc = __builtin_amdgcn_wmma_f32_16x16x32_bf16(false, a, false, bm, (short)0, acc, false, false);
  }
  const int mo = (lane & 16) ? 8 : 0;
  bf16x8 outv;
#pragma unroll
  for (int r = 0; r < 8; ++r) outv[r] = (bf16_t)(acc[r] + in_b[dtile + mo + r]);
  *(bf16x8*)(enc_bf + (size_t)n * D_CB_ + dtile + mo) = outv;
}

/* ====== VQ: argmax_k enc[n,:]·c_norm[k,:]  (== argmin of L2 distance) ======
 * 8 waves/block share each 16-code tile via double-buffered async LDS stage. */
__global__ void __launch_bounds__(256)
vq_kernel(const bf16_t* __restrict__ enc_bf, const bf16_t* __restrict__ cn_bf,
          int* __restrict__ idx_ws, float* __restrict__ idx_out) {
  __shared__ __align__(32) bf16_t btile[2][16 * LDS_PITCH_];
  const int tid = threadIdx.x;
  const int wid = tid >> 5, lane = tid & 31;
  const int ntile = (blockIdx.x * 8 + wid) << 4;
  const int lm = lane & 15;

  /* resident A fragments: 16 tokens x K=256 */
  const bf16_t* arow = enc_bf + (size_t)(ntile + lm) * D_CB_;
  v16bf afrag[8];
#pragma unroll
  for (int kc = 0; kc < 8; ++kc) afrag[kc] = load_fragA(arow, kc * 32, lane);

  float bestv[8];
  int   besti[8];
#pragma unroll
  for (int r = 0; r < 8; ++r) { bestv[r] = -3.4e38f; besti[r] = 0; }

  /* staging: 16 rows x 512B = 512 x 16B chunks; 2 chunks per thread */
  const int ch0 = tid * 2;
  const int srow = ch0 >> 5;                  /* both chunks in same row */
  const int soff = (ch0 & 31) * 16;           /* byte offset within row  */
  const int koffB = (lane & 16) ? 16 : 0;

  {
    const char* g = (const char*)cn_bf + (size_t)srow * 512 + soff;
    char* l = (char*)&btile[0][0] + (size_t)srow * (LDS_PITCH_ * 2) + soff;
    copy16_g2l(g, l);
    copy16_g2l(g + 16, l + 16);
  }
  for (int j = 0; j < K_CB_ / 16; ++j) {
    wait_async_copies();      /* our copies into btile[j&1] done   */
    __syncthreads();          /* all threads' copies done/visible; prev compute done */
    if (j + 1 < K_CB_ / 16) {
      const char* g = (const char*)cn_bf + (size_t)(j + 1) * 8192 + (size_t)srow * 512 + soff;
      char* l = (char*)&btile[(j + 1) & 1][0] + (size_t)srow * (LDS_PITCH_ * 2) + soff;
      copy16_g2l(g, l);
      copy16_g2l(g + 16, l + 16);
    }
    const bf16_t* brow = &btile[j & 1][0] + lm * LDS_PITCH_ + koffB;
    v8f acc = {};
#pragma unroll
    for (int kc = 0; kc < 8; ++kc) {
      const v16bf bm = load16_contig(brow + kc * 32);
      acc = __builtin_amdgcn_wmma_f32_16x16x32_bf16(false, afrag[kc], false, bm, (short)0, acc, false, false);
    }
    const int code = (j << 4) + lm;
#pragma unroll
    for (int r = 0; r < 8; ++r)
      if (acc[r] > bestv[r]) { bestv[r] = acc[r]; besti[r] = code; }
  }
  /* reduce over the 16 lanes of each half-wave (masks 1..8 stay in-half) */
#pragma unroll
  for (int r = 0; r < 8; ++r) {
#pragma unroll
    for (int off = 1; off < 16; off <<= 1) {
      const float ov = __shfl_xor(bestv[r], off, 32);
      const int   oi = __shfl_xor(besti[r], off, 32);
      if (ov > bestv[r] || (ov == bestv[r] && oi < besti[r])) { bestv[r] = ov; besti[r] = oi; }
    }
  }
  if (lm == 0) {
    const int mo = (lane & 16) ? 8 : 0;
#pragma unroll
    for (int r = 0; r < 8; ++r) {
      const int n = ntile + mo + r;
      idx_ws[n]  = besti[r];
      idx_out[n] = (float)besti[r];
    }
  }
}

/* ========== out_proj: out = w_out @ codebook[idx] + out_b ==================
 * 8 waves/block share one 16-token z_q tile staged (gathered) into LDS.     */
__global__ void __launch_bounds__(256)
outproj_kernel(const bf16_t* __restrict__ cb_bf, const bf16_t* __restrict__ w_out_bf,
               const int* __restrict__ idx_ws, const float* __restrict__ out_b,
               float* __restrict__ out) {
  __shared__ __align__(32) bf16_t ztile[16 * LDS_PITCH_];
  __shared__ int codes[16];
  const int tid = threadIdx.x;
  const int wid = tid >> 5, lane = tid & 31;
  const int ntile  = (blockIdx.x >> 3) << 4;               /* 2048 token tiles */
  const int ctile  = (((blockIdx.x & 7) << 3) + wid) << 4; /* 64 C-tiles       */
  const int lm = lane & 15;
  const int n  = ntile + lm;

  if (tid < 16) codes[tid] = idx_ws[ntile + tid];
  __syncthreads();

  /* gather 16 codebook rows (bf16) into LDS: 2 x 16B chunks per thread */
  {
    const int ch0 = tid * 2;
    const int r = ch0 >> 5;
    const int o = (ch0 & 31) * 16;
    const char* g = (const char*)(cb_bf + (size_t)codes[r] * D_CB_) + o;
    char* l = (char*)ztile + (size_t)r * (LDS_PITCH_ * 2) + o;
    copy16_g2l(g, l);
    copy16_g2l(g + 16, l + 16);
  }
  wait_async_copies();
  __syncthreads();

  const bf16_t* arow = w_out_bf + (size_t)(ctile + lm) * D_CB_;
  const bf16_t* brow = ztile + lm * LDS_PITCH_ + ((lane & 16) ? 16 : 0);
  v8f acc = {};
#pragma unroll
  for (int kc = 0; kc < 8; ++kc) {
    const v16bf a  = load_fragA(arow, kc * 32, lane);
    const v16bf bm = load16_contig(brow + kc * 32);
    acc = __builtin_amdgcn_wmma_f32_16x16x32_bf16(false, a, false, bm, (short)0, acc, false, false);
  }
  const int b = n >> 12, t = n & (T_ - 1);
  const int mo = (lane & 16) ? 8 : 0;
  float* ocol = out + ((size_t)b * C_IN_) * T_ + t;
#pragma unroll
  for (int r = 0; r < 8; ++r) {
    const int c = ctile + mo + r;
    ocol[(size_t)c * T_] = acc[r] + out_b[c];
  }
}

/* ============================== launcher =================================== */
extern "C" void kernel_launch(void* const* d_in, const int* in_sizes, int n_in,
                              void* d_out, int out_size, void* d_ws, size_t ws_size,
                              hipStream_t stream) {
  (void)in_sizes; (void)n_in; (void)out_size; (void)ws_size;
  const float* z        = (const float*)d_in[0];
  const float* in_v     = (const float*)d_in[1];
  const float* in_g     = (const float*)d_in[2];
  const float* in_b     = (const float*)d_in[3];
  const float* out_v    = (const float*)d_in[4];
  const float* out_g    = (const float*)d_in[5];
  const float* out_b    = (const float*)d_in[6];
  const float* codebook = (const float*)d_in[7];

  char* ws = (char*)d_ws;
  bf16_t* w_in_bf  = (bf16_t*)(ws + 0);                     /* 512 KB */
  bf16_t* w_out_bf = (bf16_t*)(ws + 524288);                /* 512 KB */
  bf16_t* cb_bf    = (bf16_t*)(ws + 1048576);               /* 4 MB   */
  bf16_t* cn_bf    = (bf16_t*)(ws + 5242880);               /* 4 MB   */
  bf16_t* enc_bf   = (bf16_t*)(ws + 9437184);               /* 16 MB  */
  int*    idx_ws   = (int*)   (ws + 26214400);              /* 128 KB */

  float* out_f   = (float*)d_out;
  float* idx_out = out_f + OUT_ELEMS_;

  prep_kernel<<<D_CB_ + C_IN_ + K_CB_, 256, 0, stream>>>(
      in_v, in_g, out_v, out_g, codebook, w_in_bf, w_out_bf, cb_bf, cn_bf);

  /* 16 D-tiles x 2048 N-tiles, 8 waves/block */
  inproj_kernel<<<(16 * 2048) / 8, 256, 0, stream>>>(z, w_in_bf, in_b, enc_bf);

  /* 2048 token tiles, 8 waves/block, LDS-shared code tiles */
  vq_kernel<<<2048 / 8, 256, 0, stream>>>(enc_bf, cn_bf, idx_ws, idx_out);

  /* 2048 token tiles x 8 C-groups, 8 waves/block, LDS-shared z_q tile */
  outproj_kernel<<<2048 * 8, 256, 0, stream>>>(cb_bf, w_out_bf, idx_ws, out_b, out_f);
}
